// SelfAttention_26551487824001
// MI455X (gfx1250) — compile-verified, matrix operand order checked by reading, not run on hardware
//
#include <hip/hip_runtime.h>

// ---- problem constants (match reference) ----
#define DIN   1024
#define DOUT  1024
#define BB    4
#define SSQ   4096
#define MTOT  (BB * SSQ)   // 16384

typedef __attribute__((ext_vector_type(16))) __bf16 v16bf;
typedef __attribute__((ext_vector_type(8)))  __bf16 bf16x8;
typedef __attribute__((ext_vector_type(8)))  float  v8f;
typedef __attribute__((ext_vector_type(4)))  unsigned int u32x4;
typedef __attribute__((ext_vector_type(8)))  unsigned int u32x8;

union Frag16 {
    v16bf  v;
    bf16x8 h[2];   // h[0] = elements 0..7, h[1] = elements 8..15
};

__device__ __forceinline__ __bf16 f32_to_bf16(float f) {
    unsigned int u = __float_as_uint(f);
    u += 0x7FFFu + ((u >> 16) & 1u);          // round-to-nearest-even
    unsigned short s = (unsigned short)(u >> 16);
    union { unsigned short u16; __bf16 b; } cv;
    cv.u16 = s;
    return cv.b;
}

// ---------------------------------------------------------------------------
// Kernel 1: fp32 x -> bf16 (coalesced, 4 elems/thread)
// ---------------------------------------------------------------------------
__global__ void cvt_x_kernel(const float* __restrict__ x, __bf16* __restrict__ xb) {
    const int base = (blockIdx.x * blockDim.x + threadIdx.x) * 4;
#pragma unroll
    for (int j = 0; j < 4; ++j) xb[base + j] = f32_to_bf16(x[base + j]);
}

// ---------------------------------------------------------------------------
// Kernel 2: Wt[z][n][k] = (bf16) W[z][k][n]   (z = q,k,v)
// ---------------------------------------------------------------------------
__global__ void cvt_wT_kernel(const float* __restrict__ Wq,
                              const float* __restrict__ Wk,
                              const float* __restrict__ Wv,
                              __bf16* __restrict__ wt3) {
    const int z = blockIdx.y;
    const float* W = (z == 0) ? Wq : (z == 1) ? Wk : Wv;
    __bf16* o = wt3 + (size_t)z * DIN * DOUT;
    const int i = blockIdx.x * blockDim.x + threadIdx.x;   // over 1M elems
    const int n = i >> 10;
    const int k = i & 1023;
    o[i] = f32_to_bf16(W[k * DOUT + n]);
}

// ---------------------------------------------------------------------------
// Kernel 3: QKV projection GEMM  out = x @ W + b   (bf16 WMMA, f32 accum)
// grid (MTOT/128, DOUT/128, 3), block 256 (8 waves).
// B-tile (128 cols x 32 k of W^T) is uniform per block -> staged into LDS by
// the Tensor Data Mover (double-buffered, TENSORcnt), issued by wave 0.
// A-fragments (per-wave x rows) prefetched one k-step ahead in registers.
// ---------------------------------------------------------------------------
__global__ void __launch_bounds__(256)
qkv_proj_kernel(const __bf16* __restrict__ xb,
                const __bf16* __restrict__ wt3,
                const float* __restrict__ bq,
                const float* __restrict__ bk,
                const float* __restrict__ bv,
                __bf16* __restrict__ qb,
                __bf16* __restrict__ kbuf,
                __bf16* __restrict__ vtb) {
    __shared__ __align__(16) __bf16 wtile[2][128][32];   // 2 x 8 KB W^T tiles

    const int lane  = threadIdx.x & 31;
    const int w     = threadIdx.x >> 5;
    const int m0    = blockIdx.x * 128;
    const int n0    = blockIdx.y * 128;
    const int z     = blockIdx.z;
    const __bf16* wt = wt3 + (size_t)z * DIN * DOUT;
    const float* bias = (z == 0) ? bq : (z == 1) ? bk : bv;

    const int col15 = lane & 15;
    const int kOff  = (lane < 16) ? 0 : 8;    // ISA 16-bit A/B fragment K offset
    const int arow  = m0 + w * 16 + col15;    // A-fragment row (lane = M)

    // ---- Tensor DMA descriptor (all-uniform values; live in SGPRs) ----
    const unsigned lds_a[2] = {
        (unsigned)(uintptr_t)&wtile[0][0][0],
        (unsigned)(uintptr_t)&wtile[1][0][0],
    };
    u32x8 g1;
    g1[0] = 0x00010000u;                              // mask=0, data_size=1 (2B)
    g1[1] = (DIN & 0xFFFFu) << 16;                    // tensor_dim0 lo16
    g1[2] = (DIN >> 16) | ((DOUT & 0xFFFFu) << 16);   // dim0 hi16 | dim1 lo16
    g1[3] = (DOUT >> 16) | (32u << 16);               // dim1 hi16 | tile_dim0=32
    g1[4] = 128u;                                     // tile_dim1=128, tile_dim2=0
    g1[5] = (unsigned)DIN;                            // tensor_dim0_stride lo32
    g1[6] = 0u;                                       // stride hi | dim1_stride lo
    g1[7] = 0u;

    v8f acc[8] = {};

    const int NSTEP = DIN / 32;   // 32

    // prologue: A-frag for step 0 + TDM of tile 0
    Frag16 a_cur, a_nxt;
    {
        const __bf16* ap = xb + (size_t)arow * DIN + kOff;
        a_cur.h[0] = *reinterpret_cast<const bf16x8*>(ap);
        a_cur.h[1] = *reinterpret_cast<const bf16x8*>(ap + 16);
    }
    if (w == 0) {
        const unsigned long long ga =
            (unsigned long long)(uintptr_t)wt + 2ull * ((unsigned long long)n0 * DIN);
        u32x4 g0;
        g0[0] = 1u;                                   // count=1, user mode
        g0[1] = lds_a[0];
        g0[2] = (unsigned)ga;
        g0[3] = ((unsigned)(ga >> 32) & 0x01FFFFFFu) | 0x80000000u;  // type=2
        asm volatile("tensor_load_to_lds %0, %1" :: "s"(g0), "s"(g1) : "memory");
    }

    for (int i = 0; i < NSTEP; ++i) {
        const int nxt = i + 1;
        if (nxt < NSTEP) {
            // prefetch next A-fragment (registers)
            const __bf16* ap = xb + (size_t)arow * DIN + nxt * 32 + kOff;
            a_nxt.h[0] = *reinterpret_cast<const bf16x8*>(ap);
            a_nxt.h[1] = *reinterpret_cast<const bf16x8*>(ap + 16);
            // kick TDM for next W^T tile into the other LDS buffer
            if (w == 0) {
                const unsigned long long ga =
                    (unsigned long long)(uintptr_t)wt +
                    2ull * ((unsigned long long)n0 * DIN + (unsigned)nxt * 32u);
                u32x4 g0;
                g0[0] = 1u;
                g0[1] = lds_a[nxt & 1];
                g0[2] = (unsigned)ga;
                g0[3] = ((unsigned)(ga >> 32) & 0x01FFFFFFu) | 0x80000000u;
                asm volatile("tensor_load_to_lds %0, %1" :: "s"(g0), "s"(g1) : "memory");
            }
            __builtin_amdgcn_s_wait_tensorcnt((short)1);  // tile i complete
        } else {
            __builtin_amdgcn_s_wait_tensorcnt((short)0);
        }
        __syncthreads();                                   // tile i visible to all

        const int bufi = i & 1;
#pragma unroll
        for (int ct = 0; ct < 8; ++ct) {
            const int c = ct * 16 + col15;                 // B-frag col (lane = N)
            Frag16 bfr;
            bfr.h[0] = *reinterpret_cast<const bf16x8*>(&wtile[bufi][c][kOff]);
            bfr.h[1] = *reinterpret_cast<const bf16x8*>(&wtile[bufi][c][kOff + 16]);
            acc[ct] = __builtin_amdgcn_wmma_f32_16x16x32_bf16(
                false, a_cur.v, false, bfr.v, (short)0, acc[ct], false, false);
        }
        __syncthreads();                                   // done reading buf i
        a_cur = a_nxt;
    }

    // epilogue: +bias, convert to bf16, store (C layout: row = e + half*8)
    const int rbase = m0 + w * 16 + ((lane < 16) ? 0 : 8);
#pragma unroll
    for (int ct = 0; ct < 8; ++ct) {
        const int col = n0 + ct * 16 + col15;
        const float bv_ = bias[col];
#pragma unroll
        for (int e = 0; e < 8; ++e) {
            const int row = rbase + e;
            const __bf16 hval = f32_to_bf16(acc[ct][e] + bv_);
            if (z == 0) {
                qb[(size_t)row * DOUT + col] = hval;
            } else if (z == 1) {
                kbuf[(size_t)row * DOUT + col] = hval;
            } else {
                const int bidx = row >> 12;           // row / 4096
                const int srow = row & (SSQ - 1);
                vtb[((size_t)bidx * DOUT + col) * SSQ + srow] = hval;
            }
        }
    }
}

// ---------------------------------------------------------------------------
// Kernel 4: flash attention. Block = 32 query rows, 8 waves split the
// 1024-dim axis (128 each). Q.K^T partials reduced via ds_add_f32 into a
// 32x32 LDS tile; wave 0 does the online softmax while the other waves'
// V-fragment loads are in flight; P goes through LDS (C-layout -> A-layout).
// ---------------------------------------------------------------------------
__global__ void __launch_bounds__(256)
attn_kernel(const __bf16* __restrict__ qb,
            const __bf16* __restrict__ kbuf,
            const __bf16* __restrict__ vtb,
            float* __restrict__ out) {
    __shared__ float sc[32][32];                      // reduced scores (f32)
    __shared__ __align__(16) __bf16 p_lds[32][32];    // softmaxed P (bf16)
    __shared__ float alpha_lds[32];
    __shared__ float linv_lds[32];

    const int lane   = threadIdx.x & 31;
    const int w      = threadIdx.x >> 5;
    const int b      = blockIdx.x >> 7;               // / (S/32)
    const int q0     = (blockIdx.x & 127) * 32;
    const int col15  = lane & 15;
    const int kOff   = (lane < 16) ? 0 : 8;
    const int half8  = (lane < 16) ? 0 : 8;
    const int dslice = w * 128;

    // Preload this wave's Q fragments (invariant over the key loop)
    Frag16 qf[2][4];
#pragma unroll
    for (int rb = 0; rb < 2; ++rb) {
        const int qrow = b * SSQ + q0 + rb * 16 + col15;
#pragma unroll
        for (int ks = 0; ks < 4; ++ks) {
            const __bf16* p = qb + (size_t)qrow * DOUT + dslice + ks * 32 + kOff;
            qf[rb][ks].h[0] = *reinterpret_cast<const bf16x8*>(p);
            qf[rb][ks].h[1] = *reinterpret_cast<const bf16x8*>(p + 16);
        }
    }

    v8f o[2][8] = {};          // O accumulators: 32 rows x 128 cols per wave
    float m_run = -3.0e38f;    // wave 0 only
    float l_run = 0.0f;

    for (int kb0 = 0; kb0 < SSQ; kb0 += 32) {
        __syncthreads();
        {   // zero the reduction tile (1024 floats / 256 threads)
            float* scf = &sc[0][0];
#pragma unroll
            for (int j = 0; j < 4; ++j) scf[threadIdx.x * 4 + j] = 0.0f;
        }
        __syncthreads();

        // partial scores over this wave's 128-dim slice: 16 WMMAs,
        // K-fragments double-buffered in registers (idx = ks*2 + kt).
        v8f st[2][2] = {};
        Frag16 kf_cur, kf_nxt;
        {
            const int key = b * SSQ + kb0 + col15;        // idx 0: kt=0
            const __bf16* p = kbuf + (size_t)key * DOUT + dslice + kOff;
            kf_cur.h[0] = *reinterpret_cast<const bf16x8*>(p);
            kf_cur.h[1] = *reinterpret_cast<const bf16x8*>(p + 16);
        }
#pragma unroll
        for (int idx = 0; idx < 8; ++idx) {
            if (idx < 7) {
                const int nks = (idx + 1) >> 1, nkt = (idx + 1) & 1;
                const int key = b * SSQ + kb0 + nkt * 16 + col15;
                const __bf16* p = kbuf + (size_t)key * DOUT + dslice + nks * 32 + kOff;
                kf_nxt.h[0] = *reinterpret_cast<const bf16x8*>(p);
                kf_nxt.h[1] = *reinterpret_cast<const bf16x8*>(p + 16);
            }
            const int ks = idx >> 1, kt = idx & 1;
#pragma unroll
            for (int rb = 0; rb < 2; ++rb) {
                st[rb][kt] = __builtin_amdgcn_wmma_f32_16x16x32_bf16(
                    false, qf[rb][ks].v, false, kf_cur.v, (short)0,
                    st[rb][kt], false, false);
            }
            kf_cur = kf_nxt;
        }

        // reduce partials across the 8 waves (ds_add_f32 atomics)
#pragma unroll
        for (int rb = 0; rb < 2; ++rb)
#pragma unroll
            for (int kt = 0; kt < 2; ++kt)
#pragma unroll
                for (int e = 0; e < 8; ++e)
                    atomicAdd(&sc[rb * 16 + e + half8][kt * 16 + col15],
                              st[rb][kt][e]);

        // Issue all V-fragment loads now: their latency hides under the
        // softmax phase (two barriers away from first use).
        Frag16 vf[8];
#pragma unroll
        for (int ct = 0; ct < 8; ++ct) {
            const int d = dslice + ct * 16 + col15;       // B-frag col = d_out
            const __bf16* vp = vtb + ((size_t)b * DOUT + d) * SSQ + kb0 + kOff;
            vf[ct].h[0] = *reinterpret_cast<const bf16x8*>(vp);
            vf[ct].h[1] = *reinterpret_cast<const bf16x8*>(vp + 16);
        }

        // hint: pull next key block toward the caches (global_prefetch_b8)
        if (kb0 + 32 < SSQ) {
            const int nkey = b * SSQ + kb0 + 32 + col15;
            __builtin_prefetch(kbuf + (size_t)nkey * DOUT + dslice, 0, 0);
        }
        __syncthreads();

        // wave 0: online softmax, one query row per lane
        if (w == 0) {
            const float rscale = 0.03125f;   // 1/sqrt(1024)
            float mb = -3.0e38f;
#pragma unroll
            for (int j = 0; j < 32; ++j) mb = fmaxf(mb, sc[lane][j]);
            const float m_new = fmaxf(m_run, mb * rscale);
            const float alpha = __expf(m_run - m_new);
            float psum = 0.0f;
#pragma unroll
            for (int j = 0; j < 32; ++j) {
                const float pv = __expf(sc[lane][j] * rscale - m_new);
                p_lds[lane][j] = f32_to_bf16(pv);
                psum += pv;
            }
            l_run = l_run * alpha + psum;
            m_run = m_new;
            alpha_lds[lane] = alpha;
        }
        __syncthreads();

        // rescale O by alpha(row), then O += P @ V  (16 WMMAs)
        float a0[8], a1[8];
#pragma unroll
        for (int e = 0; e < 8; ++e) {
            a0[e] = alpha_lds[e + half8];
            a1[e] = alpha_lds[16 + e + half8];
        }
#pragma unroll
        for (int ct = 0; ct < 8; ++ct)
#pragma unroll
            for (int e = 0; e < 8; ++e) {
                o[0][ct][e] *= a0[e];
                o[1][ct][e] *= a1[e];
            }
        Frag16 pf[2];   // A-fragments of P, read back through LDS
#pragma unroll
        for (int rb = 0; rb < 2; ++rb) {
            const __bf16* pp = &p_lds[rb * 16 + col15][kOff];
            pf[rb].h[0] = *reinterpret_cast<const bf16x8*>(pp);
            pf[rb].h[1] = *reinterpret_cast<const bf16x8*>(pp + 16);
        }
#pragma unroll
        for (int ct = 0; ct < 8; ++ct) {
#pragma unroll
            for (int rb = 0; rb < 2; ++rb)
                o[rb][ct] = __builtin_amdgcn_wmma_f32_16x16x32_bf16(
                    false, pf[rb].v, false, vf[ct].v, (short)0, o[rb][ct],
                    false, false);
        }
    }

    if (w == 0) linv_lds[lane] = 1.0f / l_run;
    __syncthreads();

    // write O / l  (fp32 output)
#pragma unroll
    for (int rb = 0; rb < 2; ++rb) {
#pragma unroll
        for (int ct = 0; ct < 8; ++ct) {
            const int col = dslice + ct * 16 + col15;
#pragma unroll
            for (int e = 0; e < 8; ++e) {
                const int rit = rb * 16 + e + half8;
                const int row = q0 + rit;
                out[((size_t)b * SSQ + row) * DOUT + col] = o[rb][ct][e] * linv_lds[rit];
            }
        }
    }
}

// ---------------------------------------------------------------------------
// Host-side launcher
// ---------------------------------------------------------------------------
extern "C" void kernel_launch(void* const* d_in, const int* in_sizes, int n_in,
                              void* d_out, int out_size, void* d_ws, size_t ws_size,
                              hipStream_t stream) {
    const float* x  = (const float*)d_in[0];
    const float* Wq = (const float*)d_in[1];
    const float* bq = (const float*)d_in[2];
    const float* Wk = (const float*)d_in[3];
    const float* bk = (const float*)d_in[4];
    const float* Wv = (const float*)d_in[5];
    const float* bv = (const float*)d_in[6];
    float* out = (float*)d_out;

    // workspace layout (bytes):
    //  xb: 32MB | wt3: 6MB | qb: 32MB | kb: 32MB | vtb: 32MB
    char* ws = (char*)d_ws;
    __bf16* xb  = (__bf16*)(ws);
    __bf16* wt3 = (__bf16*)(ws + 33554432);
    __bf16* qb  = (__bf16*)(ws + 33554432 + 6291456);
    __bf16* kb  = (__bf16*)(ws + 33554432 + 6291456 + 33554432);
    __bf16* vtb = (__bf16*)(ws + 33554432 + 6291456 + 2 * 33554432);

    cvt_x_kernel<<<16384, 256, 0, stream>>>(x, xb);                       // 16M elems
    cvt_wT_kernel<<<dim3(4096, 3), 256, 0, stream>>>(Wq, Wk, Wv, wt3);    // 3 x 1M
    qkv_proj_kernel<<<dim3(MTOT / 128, DOUT / 128, 3), 256, 0, stream>>>(
        xb, wt3, bq, bk, bv, qb, kb, vtb);
    attn_kernel<<<BB * (SSQ / 32), 256, 0, stream>>>(qb, kb, vtb, out);
}
